// BinarizedLeNet5_30511447670911
// MI455X (gfx1250) — 16-bit and 8-bit matrix operand order repaired and checked by reading, compiled, not run on hardware
//
#include <hip/hip_runtime.h>
#include <hip/hip_bf16.h>
#include <cstdint>
#include <cmath>


typedef __attribute__((ext_vector_type(8)))  int      v8i;
typedef __attribute__((ext_vector_type(8)))  float    v8f;
typedef __attribute__((ext_vector_type(16))) _Float16 v16h;

union U16H { v8i i; v16h h; };

__constant__ int IDX3c[18] = {0,1,2, 1,2,3, 2,3,4, 3,4,5, 0,4,5, 0,1,5};
__constant__ int IDX4c[36] = {0,1,2,3, 1,2,3,4, 2,3,4,5, 0,3,4,5, 0,1,4,5,
                              0,1,2,5, 0,1,3,4, 1,2,4,5, 0,2,3,5};

__global__ void k_c1_pool_sign(const float* __restrict__ x,
                               const float* __restrict__ c1w,
                               const float* __restrict__ c1b,
                               int8_t* __restrict__ s1, int B) {
  int t = blockIdx.x * blockDim.x + threadIdx.x;
  int total = B * 6 * 14 * 14;
  if (t >= total) return;
  int j = t % 14; int r = t / 14;
  int i = r % 14; r /= 14;
  int c = r % 6;  int b = r / 6;

  const float* xb = x + (size_t)b * 1024;
  const float* wc = c1w + c * 25;
  int y0 = 2 * i, x0 = 2 * j;
  float a00 = 0.f, a01 = 0.f, a10 = 0.f, a11 = 0.f;
#pragma unroll
  for (int dy = 0; dy < 5; ++dy) {
#pragma unroll
    for (int dx = 0; dx < 5; ++dx) {
      float wv = wc[dy * 5 + dx];
      const float* row0 = xb + (y0 + dy) * 32 + x0 + dx;
      const float* row1 = row0 + 32;
      a00 += wv * row0[0];
      a01 += wv * row0[1];
      a10 += wv * row1[0];
      a11 += wv * row1[1];
    }
  }
  float bv = c1b[c];
  bool any = (a00 + bv > 0.f) || (a01 + bv > 0.f) || (a10 + bv > 0.f) || (a11 + bv > 0.f);
  s1[t] = any ? 1 : 0;
}

__global__ void k_branches(const int8_t* __restrict__ s1,
                           const float* __restrict__ w3, const float* __restrict__ b3,
                           const float* __restrict__ w4, const float* __restrict__ b4,
                           const float* __restrict__ w6, const float* __restrict__ b6,
                           int8_t* __restrict__ Amat, int B) {
  int t = blockIdx.x * blockDim.x + threadIdx.x;
  int total = B * 448;
  if (t >= total) return;
  int k = t % 448;
  int b = t / 448;
  if (k >= 400) { Amat[t] = 0; return; }

  int c = k / 25;
  int p = k % 25;
  int i = p / 5, j = p % 5;

  int nch;
  int chans[6];
  const float* wp;
  float bias;
  if (c < 6) {
    nch = 3;
#pragma unroll
    for (int q = 0; q < 3; ++q) chans[q] = IDX3c[c * 3 + q];
    wp = w3 + c * 75;  bias = b3[c];
  } else if (c < 15) {
    int c4 = c - 6;
    nch = 4;
#pragma unroll
    for (int q = 0; q < 4; ++q) chans[q] = IDX4c[c4 * 4 + q];
    wp = w4 + c4 * 100; bias = b4[c4];
  } else {
    nch = 6;
#pragma unroll
    for (int q = 0; q < 6; ++q) chans[q] = q;
    wp = w6; bias = b6[0];
  }

  int a00 = 0, a01 = 0, a10 = 0, a11 = 0;
  int y0 = 2 * i, x0 = 2 * j;
  for (int q = 0; q < nch; ++q) {
    const int8_t* hq = s1 + ((size_t)b * 6 + chans[q]) * 196;
    const float*  wq = wp + q * 25;
#pragma unroll
    for (int dy = 0; dy < 5; ++dy) {
#pragma unroll
      for (int dx = 0; dx < 5; ++dx) {
        float wv = wq[dy * 5 + dx];
        int ws = (wv > 0.f) - (wv < 0.f);
        const int8_t* r0 = hq + (y0 + dy) * 14 + x0 + dx;
        const int8_t* r1 = r0 + 14;
        a00 += ws * (int)r0[0];
        a01 += ws * (int)r0[1];
        a10 += ws * (int)r1[0];
        a11 += ws * (int)r1[1];
      }
    }
  }
  bool any = ((float)a00 + bias > 0.f) || ((float)a01 + bias > 0.f) ||
             ((float)a10 + bias > 0.f) || ((float)a11 + bias > 0.f);
  Amat[t] = any ? 1 : 0;
}

__global__ void k_prep_weights(const float* __restrict__ c5w,
                               const float* __restrict__ f6w,
                               int8_t* __restrict__ W5,
                               _Float16* __restrict__ Wf6) {
  int t = blockIdx.x * blockDim.x + threadIdx.x;
  const int n5 = 128 * 448;
  if (t < n5) {
    int k = t % 448, n = t / 448;
    int8_t v = 0;
    if (n < 120 && k < 400) {
      float wv = c5w[n * 400 + k];
      v = (int8_t)((wv > 0.f) - (wv < 0.f));
    }
    W5[t] = v;
  } else {
    int u = t - n5;
    if (u < 96 * 128) {
      int k = u % 128, n = u / 128;
      float wv = (n < 84 && k < 120) ? f6w[n * 120 + k] : 0.f;
      Wf6[u] = (_Float16)wv;
    }
  }
}

__global__ void k_gemm_c5(const int8_t* __restrict__ Amat,
                          const int8_t* __restrict__ W5,
                          const float* __restrict__ c5b,
                          _Float16* __restrict__ H1, int numTiles) {
  int wave = (blockIdx.x * blockDim.x + threadIdx.x) >> 5;
  int lane = threadIdx.x & 31;
  if (wave >= numTiles) return;
  int tileM = wave >> 3;
  int tileN = wave & 7;
  int l15   = lane & 15;
  int khalf = lane >> 4;

  const int* apb = (const int*)(Amat + (size_t)(tileM * 16 + l15) * 448);
  const int* bpb = (const int*)(W5   + (size_t)(tileN * 16 + l15) * 448);

  v8i acc = {0, 0, 0, 0, 0, 0, 0, 0};
#pragma unroll
  for (int k0 = 0; k0 < 7; ++k0) {
    const int* ap = apb + k0 * 16;
    const int* bp = bpb + k0 * 16;
    int ai = khalf * 2;
    v8i a;
    a[0] = ap[ai];      a[1] = ap[ai + 1];
    a[2] = ap[4 + ai];  a[3] = ap[5 + ai];
    a[4] = ap[8 + ai];  a[5] = ap[9 + ai];
    a[6] = ap[12 + ai]; a[7] = ap[13 + ai];
    int bi = khalf * 2;
    v8i bm;
    bm[0] = bp[bi];     bm[1] = bp[bi + 1];
    bm[2] = bp[4 + bi]; bm[3] = bp[5 + bi];
    bm[4] = bp[8 + bi]; bm[5] = bp[9 + bi];
    bm[6] = bp[12 + bi]; bm[7] = bp[13 + bi];
    acc = __builtin_amdgcn_wmma_i32_16x16x64_iu8(true, a, true, bm, acc,
                                                 false, false);
  }

  int N  = tileN * 16 + l15;
  float bv = (N < 120) ? c5b[N] : 0.f;
#pragma unroll
  for (int r = 0; r < 8; ++r) {
    int M = tileM * 16 + r + 8 * khalf;
    float val = (float)acc[r] + bv;
    float s = (N < 120) ? ((val > 0.f) ? 1.f : ((val < 0.f) ? -1.f : 0.f)) : 0.f;
    H1[(size_t)M * 128 + N] = (_Float16)s;
  }
}

__global__ void k_gemm_f6(const _Float16* __restrict__ H1,
                          const _Float16* __restrict__ Wf6,
                          const float* __restrict__ f6b,
                          float* __restrict__ H2, int numTiles) {
  int wave = (blockIdx.x * blockDim.x + threadIdx.x) >> 5;
  int lane = threadIdx.x & 31;
  if (wave >= numTiles) return;
  int tileM = wave / 6;
  int tileN = wave % 6;
  int l15   = lane & 15;
  int khalf = lane >> 4;

  const int* apb = (const int*)(H1  + (size_t)(tileM * 16 + l15) * 128);
  const int* bpb = (const int*)(Wf6 + (size_t)(tileN * 16 + l15) * 128);

  v8f acc = {0.f, 0.f, 0.f, 0.f, 0.f, 0.f, 0.f, 0.f};
#pragma unroll
  for (int k0 = 0; k0 < 4; ++k0) {
    const int* ap = apb + k0 * 16;
    int ai = khalf * 8;
    U16H ua;
    ua.i[0] = ap[ai];     ua.i[1] = ap[ai + 1];
    ua.i[2] = ap[ai + 2]; ua.i[3] = ap[ai + 3];
    ua.i[4] = ap[ai + 4];     ua.i[5] = ap[ai + 4 + 1];
    ua.i[6] = ap[ai + 4 + 2]; ua.i[7] = ap[ai + 4 + 3];
    const int* bp = bpb + k0 * 16 + khalf * 8;
    U16H ub;
#pragma unroll
    for (int q = 0; q < 8; ++q) ub.i[q] = bp[q];
    acc = __builtin_amdgcn_wmma_f32_16x16x32_f16(false, ua.h, false, ub.h,
                                                 (short)0, acc, false, false);
  }

  int n = tileN * 16 + l15;
  float bv = (n < 84) ? f6b[n] : 0.f;
#pragma unroll
  for (int r = 0; r < 8; ++r) {
    int M = tileM * 16 + r + 8 * khalf;
    H2[(size_t)M * 96 + n] = tanhf((acc[r] + bv) * 1.7519f);
  }
}

__global__ void k_rbf(const float* __restrict__ H2,
                      const float* __restrict__ rbw,
                      const float* __restrict__ rbb,
                      float* __restrict__ out, int B) {
  int t = blockIdx.x * blockDim.x + threadIdx.x;
  if (t >= B * 10) return;
  int j = t % 10, b = t / 10;
  const float* h = H2 + (size_t)b * 96;
  const float* w = rbw + j * 84;
  float s = rbb[j];
#pragma unroll 4
  for (int n = 0; n < 84; ++n) s += h[n] * w[n];
  out[t] = s > 0.f ? s : 0.f;
}

extern "C" void kernel_launch(void* const* d_in, const int* in_sizes, int n_in,
                              void* d_out, int out_size, void* d_ws, size_t ws_size,
                              hipStream_t stream) {
  const float* x    = (const float*)d_in[0];
  const float* c1w  = (const float*)d_in[1];
  const float* c1b  = (const float*)d_in[2];
  const float* w3   = (const float*)d_in[3];
  const float* b3   = (const float*)d_in[4];
  const float* w4   = (const float*)d_in[5];
  const float* b4   = (const float*)d_in[6];
  const float* w6   = (const float*)d_in[7];
  const float* b6   = (const float*)d_in[8];
  const float* c5w  = (const float*)d_in[9];
  const float* c5b  = (const float*)d_in[10];
  const float* f6w  = (const float*)d_in[11];
  const float* f6b  = (const float*)d_in[12];
  const float* rbw  = (const float*)d_in[13];
  const float* rbb  = (const float*)d_in[14];
  float* out = (float*)d_out;

  const int B = in_sizes[0] / 1024;

  char* ws = (char*)d_ws;
  int8_t*   s1   = (int8_t*)ws;
  size_t off = (size_t)B * 1176;
  int8_t*   Amat = (int8_t*)(ws + off);        off += (size_t)B * 448;
  int8_t*   W5   = (int8_t*)(ws + off);        off += (size_t)128 * 448;
  _Float16* Wf6  = (_Float16*)(ws + off);      off += (size_t)96 * 128 * 2;
  _Float16* H1   = (_Float16*)(ws + off);      off += (size_t)B * 128 * 2;
  float*    H2   = (float*)(ws + off);

  const int TPB = 256;

  {
    int total = B * 6 * 14 * 14;
    k_c1_pool_sign<<<(total + TPB - 1) / TPB, TPB, 0, stream>>>(x, c1w, c1b, s1, B);
  }
  {
    int total = B * 448;
    k_branches<<<(total + TPB - 1) / TPB, TPB, 0, stream>>>(
        s1, w3, b3, w4, b4, w6, b6, Amat, B);
  }
  {
    int total = 128 * 448 + 96 * 128;
    k_prep_weights<<<(total + TPB - 1) / TPB, TPB, 0, stream>>>(c5w, f6w, W5, Wf6);
  }
  {
    int tiles = (B / 16) * 8;
    int blocks = (tiles + 7) / 8;
    k_gemm_c5<<<blocks, TPB, 0, stream>>>(Amat, W5, c5b, H1, tiles);
  }
  {
    int tiles = (B / 16) * 6;
    int blocks = (tiles + 7) / 8;
    k_gemm_f6<<<blocks, TPB, 0, stream>>>(H1, Wf6, f6b, H2, tiles);
  }
  {
    int total = B * 10;
    k_rbf<<<(total + TPB - 1) / TPB, TPB, 0, stream>>>(H2, rbw, rbb, out, B);
  }
}
